// MiniAgentBlock_12678743458187
// MI455X (gfx1250) — matrix operand order repaired and checked by reading, not compiled, not run on hardware
//
#include <hip/hip_runtime.h>
#include <hip/hip_bf16.h>

// ---------------- problem constants (match reference) ----------------
constexpr int CB  = 2;
constexpr int CS  = 2048;
constexpr int CH  = 2048;
constexpr int CNH = 16;
constexpr int CNKV= 4;
constexpr int CHD = 128;
constexpr int CFF = 5632;
constexpr int CBS = CB * CS;          // 4096 rows
constexpr float CEPS = 1e-5f;

// ---------------- WMMA fragment types (wave32, gfx1250) ----------------
typedef float  v8f   __attribute__((ext_vector_type(8)));
typedef __bf16 v16bf __attribute__((ext_vector_type(16)));

union Frag { v16bf v; uint4 u[2]; };

static __device__ __forceinline__ v8f wmma_bf16(const Frag& a, const Frag& b, v8f c) {
  // D = A(16x32 bf16) * B(32x16 bf16) + C(16x16 f32)
  return __builtin_amdgcn_wmma_f32_16x16x32_bf16(false, a.v, false, b.v,
                                                 (short)0, c, false, false);
}

static __device__ __forceinline__ __hip_bfloat16 f2bf(float f) { return __float2bfloat16(f); }
static __device__ __forceinline__ float bf2f(__hip_bfloat16 h) { return __bfloat162float(h); }

// CDNA5 async global->LDS copy (ASYNCcnt-tracked, bypasses VGPRs).
// vdst = LDS byte offset (per-lane VGPR), vaddr = 64-bit global address.
static __device__ __forceinline__ void async_b128(unsigned lds_off, const void* gptr) {
  asm volatile("global_load_async_to_lds_b128 %0, %1, off"
               :: "v"(lds_off), "v"(gptr) : "memory");
}
static __device__ __forceinline__ void async_wait0() {
  asm volatile("s_wait_asynccnt 0x0" ::: "memory");
}
static __device__ __forceinline__ unsigned lds_addr(const void* p) {
  return (unsigned)(uintptr_t)p;   // addrspace(3) pointer -> LDS byte offset
}

// =====================================================================
// Weight transpose + convert: W fp32 (K x N, row-major) -> Wt bf16 (N x K)
// One-time pass so GEMM tiles become contiguous b128 async copies.
// =====================================================================
__global__ __launch_bounds__(256) void cvt_transpose_bf16(
    const float* __restrict__ W, __hip_bfloat16* __restrict__ Wt, int K, int N) {
  __shared__ float t[32][33];
  const int n0 = blockIdx.x * 32, k0 = blockIdx.y * 32;
  #pragma unroll
  for (int i = 0; i < 4; ++i) {
    const int lin = threadIdx.x + i * 256;
    const int r = lin >> 5, c = lin & 31;
    t[r][c] = W[(size_t)(k0 + r) * N + n0 + c];
  }
  __syncthreads();
  #pragma unroll
  for (int i = 0; i < 4; ++i) {
    const int lin = threadIdx.x + i * 256;
    const int r = lin >> 5, c = lin & 31;
    Wt[(size_t)(n0 + r) * K + k0 + c] = f2bf(t[c][r]);
  }
}

// =====================================================================
// RMSNorm: fp32 row -> bf16 row (x * rsqrt(mean(x^2)+eps) * w)
// =====================================================================
__global__ __launch_bounds__(256) void rmsnorm_bf16(const float* __restrict__ x,
                                                    const float* __restrict__ w,
                                                    __hip_bfloat16* __restrict__ out) {
  const int row = blockIdx.x;
  const float* xr = x + (size_t)row * CH;
  float ss = 0.f;
  for (int i = threadIdx.x; i < CH / 4; i += blockDim.x) {
    float4 v = reinterpret_cast<const float4*>(xr)[i];
    ss += v.x * v.x + v.y * v.y + v.z * v.z + v.w * v.w;
  }
  for (int o = 16; o; o >>= 1) ss += __shfl_xor(ss, o, 32);
  __shared__ float red[8];
  if ((threadIdx.x & 31) == 0) red[threadIdx.x >> 5] = ss;
  __syncthreads();
  __shared__ float rs_sh;
  if (threadIdx.x == 0) {
    float t = 0.f;
    for (int i = 0; i < 8; ++i) t += red[i];
    rs_sh = rsqrtf(t / (float)CH + CEPS);
  }
  __syncthreads();
  const float rs = rs_sh;
  __hip_bfloat16* orow = out + (size_t)row * CH;
  for (int i = threadIdx.x; i < CH; i += blockDim.x)
    orow[i] = f2bf(xr[i] * rs * w[i]);
}

// =====================================================================
// GEMM: C(MxN) = A_bf16(MxK, row-major) * Wt_bf16(NxK, row-major = B^T).
// Block tile 128x128, 8 waves as 4(M) x 2(N), wave tile 32x64 =
// 8 x v_wmma_f32_16x16x32_bf16 per K-step of 32. Double-buffered LDS fed
// by global_load_async_to_lds_b128, synced with s_wait_asynccnt.
// MODE 0: Cf = acc + Res (fp32)        MODE 1: Cb = bf16(acc)
// MODE 2: Cb = bf16(silu(acc))         MODE 3: Cb = bf16(acc * Cb)
// =====================================================================
template<int MODE>
__global__ __launch_bounds__(256) void gemm_bf16_wmma(
    const __hip_bfloat16* __restrict__ A, const __hip_bfloat16* __restrict__ Bt,
    float* Cf, __hip_bfloat16* Cb, const float* __restrict__ Res,
    int M, int N, int K)
{
  constexpr int BM = 128, BN = 128, BK = 32;
  constexpr int LD = 48;                       // halves per LDS row (96B, 16B-aligned)
  __shared__ __align__(16) __hip_bfloat16 As[2][BM * LD];
  __shared__ __align__(16) __hip_bfloat16 Bs[2][BN * LD];

  const int tid  = threadIdx.x;
  const int gm0  = blockIdx.y * BM;
  const int gn0  = blockIdx.x * BN;
  const int w    = tid >> 5;
  const int lane = tid & 31;
  const int lh   = lane & 15;
  const int hi   = lane >> 4;
  const int wm   = w >> 1;       // 0..3
  const int wn   = w & 1;        // 0..1

  const unsigned as_base = lds_addr(&As[0][0]);
  const unsigned bs_base = lds_addr(&Bs[0][0]);
  constexpr unsigned BUFB = (unsigned)(BM * LD * 2);   // bytes per buffer

  // stage one (A,B) tile pair into buffer `buf` via async b128 copies
  auto stage = [&](int buf, int k0) {
    #pragma unroll
    for (int it = 0; it < 2; ++it) {
      const int idx = tid + it * 256;          // 0..511
      const int r   = idx >> 2;                // 0..127
      const int ch  = idx & 3;                 // 0..3 (8 halves each)
      async_b128(as_base + (unsigned)buf * BUFB + (unsigned)(r * 96 + ch * 16),
                 A  + (size_t)(gm0 + r) * K + k0 + ch * 8);
      async_b128(bs_base + (unsigned)buf * BUFB + (unsigned)(r * 96 + ch * 16),
                 Bt + (size_t)(gn0 + r) * K + k0 + ch * 8);
    }
  };

  v8f acc[2][4] = {};

  stage(0, 0);
  async_wait0();
  __syncthreads();

  int buf = 0;
  for (int k0 = 0; k0 < K; k0 += BK) {
    if (k0 + BK < K) stage(buf ^ 1, k0 + BK);   // overlap next-tile DMA with math

    // --- fragments per ISA 7.12.2 wave32 layouts ---
    Frag a[2];
    #pragma unroll
    for (int sm = 0; sm < 2; ++sm) {
      const int ar = (wm * 32 + sm * 16 + lh) * LD;
      a[sm].u[0] = *reinterpret_cast<const uint4*>(&As[buf][ar + hi * 8]);
      a[sm].u[1] = *reinterpret_cast<const uint4*>(&As[buf][ar + 16 + hi * 8]);
    }
    #pragma unroll
    for (int sn = 0; sn < 4; ++sn) {
      Frag b;
      const int bc = (wn * 64 + sn * 16 + lh) * LD + hi * 8;
      b.u[0] = *reinterpret_cast<const uint4*>(&Bs[buf][bc]);
      b.u[1] = *reinterpret_cast<const uint4*>(&Bs[buf][bc + 16]);
      #pragma unroll
      for (int sm = 0; sm < 2; ++sm)
        acc[sm][sn] = wmma_bf16(a[sm], b, acc[sm][sn]);
    }

    async_wait0();          // next tile resident before buffer flip
    __syncthreads();
    buf ^= 1;
  }

  // --- epilogue (D layout: row = hi*8 + r, col = lane%16) ---
  #pragma unroll
  for (int sm = 0; sm < 2; ++sm)
    #pragma unroll
    for (int sn = 0; sn < 4; ++sn)
      #pragma unroll
      for (int r = 0; r < 8; ++r) {
        const int row = gm0 + wm * 32 + sm * 16 + hi * 8 + r;
        const int col = gn0 + wn * 64 + sn * 16 + lh;
        const size_t off = (size_t)row * N + col;
        const float v = acc[sm][sn][r];
        if (MODE == 0) {
          Cf[off] = v + (Res ? Res[off] : 0.f);
        } else if (MODE == 1) {
          Cb[off] = f2bf(v);
        } else if (MODE == 2) {
          Cb[off] = f2bf(v / (1.f + __expf(-v)));     // silu
        } else {
          Cb[off] = f2bf(v * bf2f(Cb[off]));          // act = silu(gate)*up
        }
      }
}

// =====================================================================
// RoPE (interleaved pairs), in-place on bf16 buffer [B*S, nheads*HD]
// =====================================================================
__global__ __launch_bounds__(256) void rope_bf16(__hip_bfloat16* t,
                                                 const float* __restrict__ fc,
                                                 const float* __restrict__ fs,
                                                 int nheads) {
  const int idx   = blockIdx.x * blockDim.x + threadIdx.x;
  const int total = CBS * nheads * (CHD / 2);
  if (idx >= total) return;
  const int d2  = idx % (CHD / 2);
  const int tmp = idx / (CHD / 2);
  const int h   = tmp % nheads;
  const int bs  = tmp / nheads;
  const int s   = bs % CS;
  const float c  = fc[s * (CHD / 2) + d2];
  const float sn = fs[s * (CHD / 2) + d2];
  const size_t off = ((size_t)bs * nheads + h) * CHD + 2 * d2;
  const float t0 = bf2f(t[off]), t1 = bf2f(t[off + 1]);
  t[off]     = f2bf(t0 * c - t1 * sn);
  t[off + 1] = f2bf(t0 * sn + t1 * c);
}

// =====================================================================
// Causal GQA flash attention. Block = 4 waves x 16 q-rows = 64 q rows per
// (head, batch). kv chunks of 32: scores via 8 WMMAs, online softmax with
// width-16 shuffle reductions, P transposed through LDS, PV via 8 WMMAs
// against V staged transposed in LDS.
// =====================================================================
__global__ __launch_bounds__(128) void attn_fwd(
    const __hip_bfloat16* __restrict__ Q,
    const __hip_bfloat16* __restrict__ Kc,
    const __hip_bfloat16* __restrict__ Vc,
    __hip_bfloat16* __restrict__ O)
{
  constexpr int LDV = 48, LDP = 48;
  __shared__ __align__(16) __hip_bfloat16 Vt[CHD * LDV];       // [d][kv]  (32-kv chunk)
  __shared__ __align__(16) __hip_bfloat16 Ps[4 * 16 * LDP];    // per-wave P 16x32

  const int tid  = threadIdx.x;
  const int w    = tid >> 5;
  const int lane = tid & 31;
  const int lh   = lane & 15;
  const int hi   = lane >> 4;
  const int b    = blockIdx.z;
  const int h    = blockIdx.y;
  const int kvh  = h / (CNH / CNKV);
  const int q0b  = blockIdx.x * 64;
  const int q0w  = q0b + w * 16;
  const float scale = 0.08838834764831845f;   // 1/sqrt(128)

  Frag aq[4];
  {
    const __hip_bfloat16* qrow =
        Q + ((size_t)(b * CS + q0w + lh) * CNH + h) * CHD;
    #pragma unroll
    for (int c = 0; c < 4; ++c) {
      aq[c].u[0] = *reinterpret_cast<const uint4*>(qrow + 32 * c + hi * 8);
      aq[c].u[1] = *reinterpret_cast<const uint4*>(qrow + 32 * c + 16 + hi * 8);
    }
  }

  v8f oacc[8] = {};
  float mrow[8], lrow[8];
  #pragma unroll
  for (int r = 0; r < 8; ++r) { mrow[r] = -__builtin_inff(); lrow[r] = 0.f; }

  const int kvend = q0b + 64;
  for (int kv0 = 0; kv0 < kvend; kv0 += 32) {
    // --- cooperatively stage V chunk (32 x 128) transposed into LDS ---
    #pragma unroll
    for (int it = 0; it < 4; ++it) {
      const int idx = tid + it * 128;
      const int r   = idx >> 4;          // 0..31 kv row
      const int cg  = (idx & 15) * 8;    // 0..120 d col
      const uint4 vv = *reinterpret_cast<const uint4*>(
          Vc + ((size_t)(b * CS + kv0 + r) * CNKV + kvh) * CHD + cg);
      const __hip_bfloat16* ve = reinterpret_cast<const __hip_bfloat16*>(&vv);
      #pragma unroll
      for (int e = 0; e < 8; ++e) Vt[(cg + e) * LDV + r] = ve[e];
    }
    __syncthreads();

    if (kv0 <= q0w + 15) {     // wave has at least one unmasked column
      v8f s0 = {}, s1 = {};
      const __hip_bfloat16* kr0 =
          Kc + ((size_t)(b * CS + kv0 + lh) * CNKV + kvh) * CHD;
      const __hip_bfloat16* kr1 =
          Kc + ((size_t)(b * CS + kv0 + 16 + lh) * CNKV + kvh) * CHD;
      #pragma unroll
      for (int c = 0; c < 4; ++c) {
        Frag bk;
        bk.u[0] = *reinterpret_cast<const uint4*>(kr0 + 32 * c + hi * 8);
        bk.u[1] = *reinterpret_cast<const uint4*>(kr0 + 32 * c + 16 + hi * 8);
        s0 = wmma_bf16(aq[c], bk, s0);
        bk.u[0] = *reinterpret_cast<const uint4*>(kr1 + 32 * c + hi * 8);
        bk.u[1] = *reinterpret_cast<const uint4*>(kr1 + 32 * c + 16 + hi * 8);
        s1 = wmma_bf16(aq[c], bk, s1);
      }
      #pragma unroll
      for (int r = 0; r < 8; ++r) {
        const int row = q0w + hi * 8 + r;
        float v0 = s0[r] * scale + ((kv0 + lh)      <= row ? 0.f : -1e9f);
        float v1 = s1[r] * scale + ((kv0 + 16 + lh) <= row ? 0.f : -1e9f);
        float mx = fmaxf(v0, v1);
        for (int o = 8; o; o >>= 1) mx = fmaxf(mx, __shfl_xor(mx, o, 16));
        const float mn = fmaxf(mrow[r], mx);
        const float sc = __expf(mrow[r] - mn);
        const float p0 = __expf(v0 - mn);
        const float p1 = __expf(v1 - mn);
        float ps = p0 + p1;
        for (int o = 8; o; o >>= 1) ps += __shfl_xor(ps, o, 16);
        lrow[r] = lrow[r] * sc + ps;
        mrow[r] = mn;
        #pragma unroll
        for (int n = 0; n < 8; ++n) oacc[n][r] *= sc;
        const int prow = (w * 16 + hi * 8 + r) * LDP;
        Ps[prow + lh]      = f2bf(p0);
        Ps[prow + 16 + lh] = f2bf(p1);
      }
      asm volatile("s_wait_dscnt 0x0" ::: "memory");  // P visible wave-wide

      Frag ap;
      const int pr = (w * 16 + lh) * LDP;
      ap.u[0] = *reinterpret_cast<const uint4*>(&Ps[pr + hi * 8]);
      ap.u[1] = *reinterpret_cast<const uint4*>(&Ps[pr + 16 + hi * 8]);
      #pragma unroll
      for (int n = 0; n < 8; ++n) {
        Frag bv;
        const int vc = (n * 16 + lh) * LDV + hi * 8;
        bv.u[0] = *reinterpret_cast<const uint4*>(&Vt[vc]);
        bv.u[1] = *reinterpret_cast<const uint4*>(&Vt[vc + 16]);
        oacc[n] = wmma_bf16(ap, bv, oacc[n]);
      }
    }
    __syncthreads();
  }

  #pragma unroll
  for (int r = 0; r < 8; ++r) {
    const int row = q0w + hi * 8 + r;
    const float inv = 1.f / lrow[r];
    __hip_bfloat16* op = O + ((size_t)(b * CS + row) * CNH + h) * CHD;
    #pragma unroll
    for (int n = 0; n < 8; ++n) op[n * 16 + lh] = f2bf(oacc[n][r] * inv);
  }
}

// =====================================================================
// host orchestration
// =====================================================================
extern "C" void kernel_launch(void* const* d_in, const int* in_sizes, int n_in,
                              void* d_out, int out_size, void* d_ws, size_t ws_size,
                              hipStream_t stream) {
  (void)in_sizes; (void)n_in; (void)out_size; (void)ws_size;
  const float* x     = (const float*)d_in[0];
  const float* fcos  = (const float*)d_in[1];
  const float* fsin  = (const float*)d_in[2];
  /* d_in[3] = mask (unused; causal mask computed analytically) */
  const float* Wq    = (const float*)d_in[4];
  const float* Wk    = (const float*)d_in[5];
  const float* Wv    = (const float*)d_in[6];
  const float* Wo    = (const float*)d_in[7];
  const float* Wgate = (const float*)d_in[8];
  const float* Wup   = (const float*)d_in[9];
  const float* Wdown = (const float*)d_in[10];
  const float* w_n1  = (const float*)d_in[11];
  const float* w_n2  = (const float*)d_in[12];
  float* out = (float*)d_out;

  char* ws = (char*)d_ws;
  auto alloc = [&](size_t bytes) -> char* {
    char* p = ws; ws += (bytes + 255) & ~(size_t)255; return p;
  };
  // activations
  __hip_bfloat16* h_bf  = (__hip_bfloat16*)alloc((size_t)CBS * CH * 2);
  __hip_bfloat16* qb    = (__hip_bfloat16*)alloc((size_t)CBS * CNH * CHD * 2);
  __hip_bfloat16* kb    = (__hip_bfloat16*)alloc((size_t)CBS * CNKV * CHD * 2);
  __hip_bfloat16* vb    = (__hip_bfloat16*)alloc((size_t)CBS * CNKV * CHD * 2);
  __hip_bfloat16* attnb = (__hip_bfloat16*)alloc((size_t)CBS * CNH * CHD * 2);
  float*          x2    = (float*)         alloc((size_t)CBS * CH * 4);
  __hip_bfloat16* h2b   = (__hip_bfloat16*)alloc((size_t)CBS * CH * 2);
  __hip_bfloat16* actb  = (__hip_bfloat16*)alloc((size_t)CBS * CFF * 2);
  // pre-transposed bf16 weights (N x K each)
  __hip_bfloat16* Wq_t  = (__hip_bfloat16*)alloc((size_t)CH * CNH * CHD * 2);
  __hip_bfloat16* Wk_t  = (__hip_bfloat16*)alloc((size_t)CH * CNKV * CHD * 2);
  __hip_bfloat16* Wv_t  = (__hip_bfloat16*)alloc((size_t)CH * CNKV * CHD * 2);
  __hip_bfloat16* Wo_t  = (__hip_bfloat16*)alloc((size_t)CNH * CHD * CH * 2);
  __hip_bfloat16* Wg_t  = (__hip_bfloat16*)alloc((size_t)CH * CFF * 2);
  __hip_bfloat16* Wu_t  = (__hip_bfloat16*)alloc((size_t)CH * CFF * 2);
  __hip_bfloat16* Wd_t  = (__hip_bfloat16*)alloc((size_t)CFF * CH * 2);

  const dim3 blk256(256), blk128(128);

  // 0) weight transpose+convert (fp32 KxN -> bf16 NxK), one pass each
  cvt_transpose_bf16<<<dim3((CNH*CHD)/32, CH/32),  blk256, 0, stream>>>(Wq,    Wq_t, CH,  CNH*CHD);
  cvt_transpose_bf16<<<dim3((CNKV*CHD)/32, CH/32), blk256, 0, stream>>>(Wk,    Wk_t, CH,  CNKV*CHD);
  cvt_transpose_bf16<<<dim3((CNKV*CHD)/32, CH/32), blk256, 0, stream>>>(Wv,    Wv_t, CH,  CNKV*CHD);
  cvt_transpose_bf16<<<dim3(CH/32, (CNH*CHD)/32),  blk256, 0, stream>>>(Wo,    Wo_t, CNH*CHD, CH);
  cvt_transpose_bf16<<<dim3(CFF/32, CH/32),        blk256, 0, stream>>>(Wgate, Wg_t, CH,  CFF);
  cvt_transpose_bf16<<<dim3(CFF/32, CH/32),        blk256, 0, stream>>>(Wup,   Wu_t, CH,  CFF);
  cvt_transpose_bf16<<<dim3(CH/32, CFF/32),        blk256, 0, stream>>>(Wdown, Wd_t, CFF, CH);

  // 1) h = rmsnorm(x, w_norm1) -> bf16
  rmsnorm_bf16<<<CBS, blk256, 0, stream>>>(x, w_n1, h_bf);

  // 2) q/k/v projections (bf16 out)
  gemm_bf16_wmma<1><<<dim3((CNH * CHD) / 128, CBS / 128), blk256, 0, stream>>>(
      h_bf, Wq_t, nullptr, qb, nullptr, CBS, CNH * CHD, CH);
  gemm_bf16_wmma<1><<<dim3((CNKV * CHD) / 128, CBS / 128), blk256, 0, stream>>>(
      h_bf, Wk_t, nullptr, kb, nullptr, CBS, CNKV * CHD, CH);
  gemm_bf16_wmma<1><<<dim3((CNKV * CHD) / 128, CBS / 128), blk256, 0, stream>>>(
      h_bf, Wv_t, nullptr, vb, nullptr, CBS, CNKV * CHD, CH);

  // 3) RoPE on q and k (in place)
  {
    int tq = CBS * CNH * (CHD / 2);
    rope_bf16<<<(tq + 255) / 256, blk256, 0, stream>>>(qb, fcos, fsin, CNH);
    int tk = CBS * CNKV * (CHD / 2);
    rope_bf16<<<(tk + 255) / 256, blk256, 0, stream>>>(kb, fcos, fsin, CNKV);
  }

  // 4) causal GQA attention
  attn_fwd<<<dim3(CS / 64, CNH, CB), blk128, 0, stream>>>(qb, kb, vb, attnb);

  // 5) x2 = x + attn @ Wo  (fp32 residual)
  gemm_bf16_wmma<0><<<dim3(CH / 128, CBS / 128), blk256, 0, stream>>>(
      attnb, Wo_t, x2, nullptr, x, CBS, CH, CNH * CHD);

  // 6) h2 = rmsnorm(x2, w_norm2) -> bf16
  rmsnorm_bf16<<<CBS, blk256, 0, stream>>>(x2, w_n2, h2b);

  // 7) SwiGLU: act = silu(h2@Wgate) * (h2@Wup), fused into GEMM epilogues
  gemm_bf16_wmma<2><<<dim3(CFF / 128, CBS / 128), blk256, 0, stream>>>(
      h2b, Wg_t, nullptr, actb, nullptr, CBS, CFF, CH);
  gemm_bf16_wmma<3><<<dim3(CFF / 128, CBS / 128), blk256, 0, stream>>>(
      h2b, Wu_t, nullptr, actb, nullptr, CBS, CFF, CH);

  // 8) out = x2 + act @ Wdown
  gemm_bf16_wmma<0><<<dim3(CH / 128, CBS / 128), blk256, 0, stream>>>(
      actb, Wd_t, out, nullptr, x2, CBS, CH, CFF);
}
